// PFPEncoder_42949673333
// MI455X (gfx1250) — matrix operand order repaired and checked by reading, not compiled, not run on hardware
//
#include <hip/hip_runtime.h>

#define NN    50000
#define NE    200000
#define NG    64
#define NDIM  75
#define BDIM  13
#define IND   32
#define INTR  128
#define FPD   4096
#define DEPTH 3

typedef __attribute__((ext_vector_type(16))) __bf16 v16bf;
typedef __attribute__((ext_vector_type(8)))  __bf16 v8bf;
typedef __attribute__((ext_vector_type(8)))  float  v8f;

static __device__ __forceinline__ unsigned short f2bf(float f) {
    unsigned int u = __float_as_uint(f);
    unsigned int r = u + 0x7FFFu + ((u >> 16) & 1u);   // round-to-nearest-even
    return (unsigned short)(r >> 16);
}
static __device__ __forceinline__ float bf2f(unsigned short u) {
    return __uint_as_float(((unsigned int)u) << 16);
}

// gfx1250 async copy: 16B global -> LDS, tracked by ASYNCcnt (ISA 15.18.3).
// Low 32 bits of a generic pointer into __shared__ are the LDS byte address
// (shared aperture is selected purely by addr[63:32]).
static __device__ __forceinline__ void async_copy16(void* lds_dst, const void* gsrc) {
    unsigned lds = (unsigned)(unsigned long long)lds_dst;
    unsigned long long ga = (unsigned long long)gsrc;
    asm volatile("global_load_async_to_lds_b128 %0, %1, off"
                 :: "v"(lds), "v"(ga) : "memory");
}
static __device__ __forceinline__ void async_wait0() {
    asm volatile("s_wait_asynccnt 0x0" ::: "memory");
}

// A fragment, 16x32 bf16 row-major tile (contiguous k order, not the ISA table order; B uses the same):
// lanes 0-15: M=lane, K=kOff+0..15
// lanes 16-31: M=lane-16, K=kOff+16..31
static __device__ __forceinline__ v16bf load_a16x32(const unsigned short* tile,
                                                    int rowStride, int lane, int kOff) {
    int m  = lane & 15;
    int k0 = kOff + ((lane < 16) ? 0  : 16);
    int k1 = kOff + ((lane < 16) ? 8  : 24);
    v8bf lo = *reinterpret_cast<const v8bf*>(tile + m * rowStride + k0);
    v8bf hi = *reinterpret_cast<const v8bf*>(tile + m * rowStride + k1);
    v16bf a;
#pragma unroll
    for (int i = 0; i < 8; ++i) { a[i] = lo[i]; a[i + 8] = hi[i]; }
    return a;
}

// B fragment, 32x16 bf16 from k-contiguous (transposed) weights W[n][k]:
// lanes 0-15: N=lane, K=kOff+0..15 ; lanes 16-31: N=lane-16, K=kOff+16..31
static __device__ __forceinline__ v16bf load_b32x16(const unsigned short* w,
                                                    int kStride, int nBase, int lane, int kOff) {
    int n = nBase + (lane & 15);
    int k = kOff + ((lane < 16) ? 0 : 16);
    const v8bf* p = reinterpret_cast<const v8bf*>(w + (long)n * kStride + k);
    v8bf lo = p[0], hi = p[1];
    v16bf b;
#pragma unroll
    for (int i = 0; i < 8; ++i) { b[i] = lo[i]; b[i + 8] = hi[i]; }
    return b;
}

// ---------------- utility kernels ----------------

__global__ void k_zero(float* p, int n) {
    int i = blockIdx.x * blockDim.x + threadIdx.x;
    if (i < n) p[i] = 0.f;
}

// Wt[n*K + k] = bf16(W[k*N + n])  (k-contiguous for WMMA B fragments)
__global__ void k_transpose_bf16(const float* __restrict__ W, unsigned short* __restrict__ Wt,
                                 int K, int Nn) {
    int i = blockIdx.x * blockDim.x + threadIdx.x;
    if (i < K * Nn) {
        int n = i / K, k = i % K;
        Wt[i] = f2bf(W[(long)k * Nn + n]);
    }
}

// h = relu(edge_attr @ We1 + be1) -> bf16 [E,128]; 2 edges per 256-thr block
__global__ void k_edge_hidden(const float* __restrict__ ea, const float* __restrict__ We1,
                              const float* __restrict__ be1, unsigned short* __restrict__ h) {
    __shared__ float w[BDIM * INTR];
    __shared__ float b[INTR];
    __shared__ float a[2 * BDIM];
    int t = threadIdx.x;
    for (int i = t; i < BDIM * INTR; i += 256) w[i] = We1[i];
    if (t < INTR) b[t] = be1[t];
    long e0 = (long)blockIdx.x * 2;
    if (t < 2 * BDIM) a[t] = ea[e0 * BDIM + t];
    __syncthreads();
    int el = t >> 7, j = t & 127;
    float acc = b[j];
#pragma unroll
    for (int k = 0; k < BDIM; ++k) acc += a[el * BDIM + k] * w[k * INTR + j];
    h[(e0 + el) * INTR + j] = f2bf(fmaxf(acc, 0.f));
}

// edge_w = h @ We2 + be2 -> bf16 [E,1024]; one 16-edge M-tile per block,
// 8 waves x 8 N-tiles, K=128 as 4 accumulating bf16 WMMAs.
// h tile (16x128 bf16 = 4 KB, contiguous) staged via async-to-LDS B128.
__global__ void __launch_bounds__(256) k_edge_w(const unsigned short* __restrict__ h,
                                                const unsigned short* __restrict__ We2t,
                                                const float* __restrict__ be2,
                                                unsigned short* __restrict__ ew) {
    __shared__ __align__(16) unsigned short hs[16 * INTR];   // 4 KB
    int t = threadIdx.x, lane = t & 31, wv = t >> 5;
    long mt = blockIdx.x;
    // one async B128 per thread: 256 * 16B = 4 KB tile, fully contiguous
    async_copy16(&hs[t * 8], h + mt * (16 * INTR) + t * 8);
    async_wait0();
    __syncthreads();
    for (int s = 0; s < 8; ++s) {
        int nt = wv + s * 8;
        v8f c = {};
#pragma unroll
        for (int kb = 0; kb < 4; ++kb) {
            v16bf a = load_a16x32(hs, INTR, lane, kb * 32);
            v16bf b = load_b32x16(We2t, INTR, nt * 16, lane, kb * 32);
            c = __builtin_amdgcn_wmma_f32_16x16x32_bf16(false, a, false, b,
                                                        (short)0, c, false, false);
        }
        int n = nt * 16 + (lane & 15);
        float bv = be2[n];
        int mBase = (lane < 16) ? 0 : 8;
#pragma unroll
        for (int r = 0; r < 8; ++r) {
            long m = mt * 16 + mBase + r;
            ew[m * 1024 + n] = f2bf(c[r] + bv);
        }
    }
}

// message0 = relu(x @ W_ae + b_ae); 8 nodes/block, lane = output channel.
// Writes both f32 message and bf16 mirror (WMMA A operand for fingerprint).
__global__ void k_atom_expand(const float* __restrict__ x, const float* __restrict__ W,
                              const float* __restrict__ b, float* __restrict__ msg,
                              unsigned short* __restrict__ msgbf) {
    __shared__ float ws_[NDIM * IND];
    __shared__ float bs[IND];
    int t = threadIdx.x;
    for (int i = t; i < NDIM * IND; i += 256) ws_[i] = W[i];
    if (t < IND) bs[t] = b[t];
    __syncthreads();
    long node = (long)blockIdx.x * 8 + (t >> 5);
    int o = t & 31;
    const float* xr = x + node * NDIM;
    float acc = bs[o];
    for (int k = 0; k < NDIM; ++k) acc += xr[k] * ws_[k * IND + o];
    float v = fmaxf(acc, 0.f);
    msg[node * IND + o] = v;
    msgbf[node * IND + o] = f2bf(v);
}

// per-edge 1x32 . 32x32 matvec, scatter-add to aggr[dst]; one wave per edge
__global__ void k_msg_scatter(const int* __restrict__ ei, const float* __restrict__ msg,
                              const unsigned short* __restrict__ ew, float* __restrict__ aggr) {
    int t = threadIdx.x;
    long e = (long)blockIdx.x * 8 + (t >> 5);
    int o = t & 31;
    int src = ei[e], dst = ei[NE + e];
    float mi = msg[(long)src * IND + o];               // lane o holds m[o]
    const unsigned short* w = ew + e * 1024;
    __builtin_prefetch(w + 1024, 0, 0);                // global_prefetch_b8: next edge
    float acc = 0.f;
#pragma unroll
    for (int i = 0; i < IND; ++i) {
        float m = __shfl(mi, i, 32);                   // broadcast m[i] across wave32
        acc += m * bf2f(w[i * IND + o]);
    }
    atomicAdd(&aggr[(long)dst * IND + o], acc);
}

// message = relu(aggr + message @ root + bias); also writes bf16 mirror
__global__ void k_node_update(const float* __restrict__ msgIn, const float* __restrict__ aggr,
                              const float* __restrict__ root, const float* __restrict__ bias,
                              float* __restrict__ msgOut, unsigned short* __restrict__ msgOutBf) {
    __shared__ float rs[IND * IND];
    __shared__ float bs[IND];
    int t = threadIdx.x;
    for (int i = t; i < IND * IND; i += 256) rs[i] = root[i];
    if (t < IND) bs[t] = bias[t];
    __syncthreads();
    long node = (long)blockIdx.x * 8 + (t >> 5);
    int o = t & 31;
    float mi = msgIn[node * IND + o];
    float acc = aggr[node * IND + o] + bs[o];
#pragma unroll
    for (int i = 0; i < IND; ++i) acc += __shfl(mi, i, 32) * rs[i * IND + o];
    float v = fmaxf(acc, 0.f);
    msgOut[node * IND + o] = v;
    msgOutBf[node * IND + o] = f2bf(v);
}

// expanded = relu(message @ Wf + bf); segment-sum over sorted batch into d_out.
// One WMMA (K=32) per 16x16 tile; A tile (16x32 bf16 = 1 KB, contiguous)
// staged via async-to-LDS B128. Uniform-graph tiles: in-register column
// reduction over M (8 accums + cross-half-wave shfl_xor) -> 16 atomics/tile.
__global__ void __launch_bounds__(256) k_fingerprint(const unsigned short* __restrict__ msgbf,
                                                     const unsigned short* __restrict__ Wft,
                                                     const float* __restrict__ bf_,
                                                     const int* __restrict__ batch,
                                                     float* __restrict__ out) {
    __shared__ __align__(16) unsigned short as_[16 * IND];  // 1 KB bf16 A tile
    __shared__ int gs[16];
    int t = threadIdx.x, lane = t & 31, wv = t >> 5;
    long mt = blockIdx.x;
    if (t < 64)                                         // 64 * 16B = 1 KB tile
        async_copy16(&as_[t * 8], msgbf + mt * (16 * IND) + t * 8);
    if (t < 16) gs[t] = batch[mt * 16 + t];
    async_wait0();
    __syncthreads();
    bool uniform = (gs[0] == gs[15]);                   // batch is sorted
    int nt = blockIdx.y * 8 + wv;
    v16bf a = load_a16x32(as_, IND, lane, 0);
    v16bf b = load_b32x16(Wft, IND, nt * 16, lane, 0);
    v8f c = {};
    c = __builtin_amdgcn_wmma_f32_16x16x32_bf16(false, a, false, b,
                                                (short)0, c, false, false);
    int n = nt * 16 + (lane & 15);
    float bv = bf_[n];
    int mBase = (lane < 16) ? 0 : 8;
    float vals[8];
#pragma unroll
    for (int r = 0; r < 8; ++r) vals[r] = fmaxf(c[r] + bv, 0.f);
    if (uniform) {
        float s = 0.f;
#pragma unroll
        for (int r = 0; r < 8; ++r) s += vals[r];       // sum M=mBase..mBase+7
        s += __shfl_xor(s, 16, 32);                     // + other half-wave's M
        if (lane < 16) atomicAdd(&out[(long)gs[0] * FPD + nt * 16 + lane], s);
    } else {
#pragma unroll
        for (int r = 0; r < 8; ++r) {
            int g = gs[mBase + r];
            atomicAdd(&out[(long)g * FPD + n], vals[r]);
        }
    }
}

// ---------------- launcher ----------------

extern "C" void kernel_launch(void* const* d_in, const int* in_sizes, int n_in,
                              void* d_out, int out_size, void* d_ws, size_t ws_size,
                              hipStream_t stream) {
    (void)in_sizes; (void)n_in; (void)out_size; (void)ws_size;
    const float* x     = (const float*)d_in[0];
    const int*   ei    = (const int*)  d_in[1];
    const float* ea    = (const float*)d_in[2];
    const int*   batch = (const int*)  d_in[3];
    const float* W_ae  = (const float*)d_in[4];
    const float* b_ae  = (const float*)d_in[5];
    const float* We1   = (const float*)d_in[6];
    const float* be1   = (const float*)d_in[7];
    const float* We2   = (const float*)d_in[8];
    const float* be2   = (const float*)d_in[9];
    const float* root  = (const float*)d_in[10];
    const float* bias  = (const float*)d_in[11];
    const float* Wf    = (const float*)d_in[12];
    const float* bfv   = (const float*)d_in[13];
    float* out = (float*)d_out;
    char*  ws  = (char*)d_ws;

    unsigned short* h_bf   = (unsigned short*)(ws);                    //  51,200,000 B
    unsigned short* edge_w = (unsigned short*)(ws + 51200000ull);      // 409,600,000 B
    unsigned short* We2t   = (unsigned short*)(ws + 460800000ull);     //     262,144 B
    unsigned short* Wft    = (unsigned short*)(ws + 461062144ull);     //     262,144 B
    float*          msgA   = (float*)(ws + 461324288ull);              //   6,400,000 B
    float*          msgB   = (float*)(ws + 467724288ull);              //   6,400,000 B
    float*          aggr   = (float*)(ws + 474124288ull);              //   6,400,000 B
    unsigned short* msgAbf = (unsigned short*)(ws + 480524288ull);     //   3,200,000 B
    unsigned short* msgBbf = (unsigned short*)(ws + 483724288ull);     //   3,200,000 B

    k_zero<<<(NG * FPD + 255) / 256, 256, 0, stream>>>(out, NG * FPD);
    k_transpose_bf16<<<(INTR * 1024 + 255) / 256, 256, 0, stream>>>(We2, We2t, INTR, 1024);
    k_transpose_bf16<<<(IND * FPD + 255) / 256, 256, 0, stream>>>(Wf, Wft, IND, FPD);
    k_edge_hidden<<<NE / 2, 256, 0, stream>>>(ea, We1, be1, h_bf);
    k_edge_w<<<NE / 16, 256, 0, stream>>>(h_bf, We2t, be2, edge_w);
    k_atom_expand<<<NN / 8, 256, 0, stream>>>(x, W_ae, b_ae, msgA, msgAbf);

    float* mIn = msgA;           float* mOut = msgB;
    unsigned short* mInBf = msgAbf; unsigned short* mOutBf = msgBbf;
    for (int d = 0; d < DEPTH; ++d) {
        k_zero<<<(NN * IND + 255) / 256, 256, 0, stream>>>(aggr, NN * IND);
        k_msg_scatter<<<NE / 8, 256, 0, stream>>>(ei, mIn, edge_w, aggr);
        k_node_update<<<NN / 8, 256, 0, stream>>>(mIn, aggr, root, bias, mOut, mOutBf);
        k_fingerprint<<<dim3(NN / 16, FPD / 128), 256, 0, stream>>>(mOutBf, Wft, bfv, batch, out);
        float* tf = mIn; mIn = mOut; mOut = tf;
        unsigned short* tb = mInBf; mInBf = mOutBf; mOutBf = tb;
    }
}
